// BrainGNN_27358941675587
// MI455X (gfx1250) — matrix operand order repaired and checked by reading, not compiled, not run on hardware
//
#include <hip/hip_runtime.h>
#include <hip/hip_bf16.h>
#include <math.h>

#define BB   512
#define RR   200
#define CC1  32
#define CC2  32
#define CC3  512
#define KKD  8
#define KK1  100
#define KK2  50
#define NEGV   (-1000000000.0f)
#define EDGEP  0.1f
#define LEAKV  0.01f
#define BNEPS  1e-5f

typedef __attribute__((ext_vector_type(16))) _Float16 v16h;
typedef __attribute__((ext_vector_type(8)))  _Float16 v8h;
typedef __attribute__((ext_vector_type(8)))  float    v8f;

__device__ __forceinline__ v16h cat8(v8h lo, v8h hi) {
  return __builtin_shufflevector(lo, hi, 0, 1, 2, 3, 4, 5, 6, 7, 8, 9, 10, 11, 12, 13, 14, 15);
}

// ---- WMMA tile loaders (CDNA5 ISA 7.12.2 layouts, f16, wave32) -------------
// A 16x32 (MxK) from row-major LDS [rows][ld]: per lane two contiguous 8-half
// runs at k0+8h and k0+16+8h -> 2x ds_load_b128.
__device__ __forceinline__ v16h lds_load_a(const _Float16* base, int ld, int r0, int k0) {
  const int lane = threadIdx.x & 31;
  const int m = lane & 15, h = lane >> 4;
  const _Float16* p = base + (r0 + m) * ld + k0 + 8 * h;
  v8h lo = *(const v8h*)p;
  v8h hi = *(const v8h*)(p + 16);
  return cat8(lo, hi);
}
// B 32x16 (KxN) from K-major (transposed) LDS BT[n][ldk]: per lane two
// contiguous 8-half runs at k0+8h and k0+16+8h -> 2x ds_load_b128.
__device__ __forceinline__ v16h lds_load_bT(const _Float16* baseT, int ldk, int k0, int n0) {
  const int lane = threadIdx.x & 31;
  const int n = lane & 15, h = lane >> 4;
  const _Float16* p = baseT + (n0 + n) * ldk + k0 + 8 * h;
  v8h lo = *(const v8h*)p;
  v8h hi = *(const v8h*)(p + 16);
  return cat8(lo, hi);
}
// C/D f32 16x16: VGPR v -> (M = v + 8*(lane>>4), N = lane&15)

// ---- 1. weight tables ------------------------------------------------------
__global__ void bg_tables(const float* __restrict__ W1a, const float* __restrict__ b1a,
                          const float* __restrict__ W2a, const float* __restrict__ b2a,
                          const float* __restrict__ W1b, const float* __restrict__ b1b,
                          const float* __restrict__ W2b, const float* __restrict__ b2b,
                          float* __restrict__ wtab1, float* __restrict__ wtab2) {
  const int T1 = RR * RR * CC1;
  const int T2 = RR * CC1 * CC2;
  int idx = blockIdx.x * blockDim.x + threadIdx.x;
  if (idx < T1) {
    int r = idx / (RR * CC1), j = idx % (RR * CC1);
    float acc = b2a[j];
#pragma unroll
    for (int k = 0; k < KKD; ++k) {
      float h = W1a[k * RR + r] + b1a[k];
      h = h > 0.f ? h : 0.f;
      acc += h * W2a[j * KKD + k];
    }
    wtab1[idx] = acc;
  } else if (idx < T1 + T2) {
    int id2 = idx - T1;
    int r = id2 / (CC1 * CC2), j = id2 % (CC1 * CC2);
    float acc = b2b[j];
#pragma unroll
    for (int k = 0; k < KKD; ++k) {
      float h = W1b[k * RR + r] + b1b[k];
      h = h > 0.f ? h : 0.f;
      acc += h * W2b[j * KKD + k];
    }
    wtab2[id2] = acc;
  }
}

// ---- 2. xt1[b,n,:] = x[b,n,:] @ wtab1[n]  (WMMA; wave = batch tile) --------
__global__ __launch_bounds__(64) void bg_xt1(const float* __restrict__ x,
                                             const float* __restrict__ wtab1,
                                             float* __restrict__ xt1) {
  const int bx = blockIdx.x;   // 0..15, two batch tiles per block
  const int n  = blockIdx.y;   // node
  const int t  = threadIdx.x;
  const int wave = t >> 5, lane = t & 31;
  __shared__ _Float16 BwT[32 * 224];      // wtab1[n] transposed, K-major, zero-padded
  __shared__ float    Dst[2 * 16 * 32];   // staged D tiles for coalesced stores
  for (int i = t; i < 224 * 32; i += 64) {
    int r = i >> 5, c = i & 31;
    BwT[c * 224 + r] = (r < RR) ? (_Float16)wtab1[(n * RR + r) * CC1 + c] : (_Float16)0.f;
  }
  __syncthreads();
  const int m = lane & 15, h = lane >> 4;
  const int brow = (bx * 2 + wave) * 16 + m;
  const float* xrow = x + ((size_t)brow * RR + n) * RR;
  v8f acc0 = {}, acc1 = {};
  for (int kt = 0; kt < 6; ++kt) {     // guard-free full tiles (k < 192)
    const int k0 = kt * 32;
    __builtin_prefetch(xrow + k0 + 64, 0, 0);
    float4 q0 = *(const float4*)(xrow + k0 + 8 * h);
    float4 q1 = *(const float4*)(xrow + k0 + 8 * h + 4);
    float4 q2 = *(const float4*)(xrow + k0 + 16 + 8 * h);
    float4 q3 = *(const float4*)(xrow + k0 + 16 + 8 * h + 4);
    v16h a;
    a[0] = (_Float16)q0.x; a[1] = (_Float16)q0.y; a[2] = (_Float16)q0.z; a[3] = (_Float16)q0.w;
    a[4] = (_Float16)q1.x; a[5] = (_Float16)q1.y; a[6] = (_Float16)q1.z; a[7] = (_Float16)q1.w;
    a[8] = (_Float16)q2.x; a[9] = (_Float16)q2.y; a[10] = (_Float16)q2.z; a[11] = (_Float16)q2.w;
    a[12] = (_Float16)q3.x; a[13] = (_Float16)q3.y; a[14] = (_Float16)q3.z; a[15] = (_Float16)q3.w;
    v16h b0 = lds_load_bT(BwT, 224, k0, 0);
    v16h b1 = lds_load_bT(BwT, 224, k0, 16);
    acc0 = __builtin_amdgcn_wmma_f32_16x16x32_f16(false, a, false, b0, (short)0, acc0, false, false);
    acc1 = __builtin_amdgcn_wmma_f32_16x16x32_f16(false, a, false, b1, (short)0, acc1, false, false);
  }
  { // tail tile k0=192: only lanes h==0 carry valid K (192..199)
    v16h a = {};
    if (h == 0) {
      float4 q0 = *(const float4*)(xrow + 192);
      float4 q1 = *(const float4*)(xrow + 196);
      a[0] = (_Float16)q0.x; a[1] = (_Float16)q0.y; a[2] = (_Float16)q0.z; a[3] = (_Float16)q0.w;
      a[4] = (_Float16)q1.x; a[5] = (_Float16)q1.y; a[6] = (_Float16)q1.z; a[7] = (_Float16)q1.w;
    }
    v16h b0 = lds_load_bT(BwT, 224, 192, 0);
    v16h b1 = lds_load_bT(BwT, 224, 192, 16);
    acc0 = __builtin_amdgcn_wmma_f32_16x16x32_f16(false, a, false, b0, (short)0, acc0, false, false);
    acc1 = __builtin_amdgcn_wmma_f32_16x16x32_f16(false, a, false, b1, (short)0, acc1, false, false);
  }
  const int colb = lane & 15;
#pragma unroll
  for (int v = 0; v < 8; ++v) {
    int M = v + 8 * h;
    Dst[(wave * 16 + M) * 32 + colb] = acc0[v];
    Dst[(wave * 16 + M) * 32 + 16 + colb] = acc1[v];
  }
  __syncthreads();
  {
    int r = t >> 1, half = t & 1;
    int bo = bx * 32 + r;
    const float* src = Dst + r * 32 + half * 16;
    float* dstp = xt1 + ((size_t)bo * RR + n) * CC1 + half * 16;
    *(float4*)(dstp + 0)  = *(const float4*)(src + 0);
    *(float4*)(dstp + 4)  = *(const float4*)(src + 4);
    *(float4*)(dstp + 8)  = *(const float4*)(src + 8);
    *(float4*)(dstp + 12) = *(const float4*)(src + 12);
  }
}

// ---- 3. h1 = softmax(masked adj) @ xt1 + bias1  (fused, WMMA) --------------
__global__ __launch_bounds__(64) void bg_h1(const float* __restrict__ adj,
                                            const float* __restrict__ xt1,
                                            const float* __restrict__ bias1,
                                            float* __restrict__ h1) {
  const int it = blockIdx.x;   // 0..12 row tile
  const int b  = blockIdx.y;
  const int t  = threadIdx.x;
  const int i0 = it * 16;
  __shared__ _Float16 Att[16 * 224];
  __shared__ _Float16 XtT[32 * 224];   // xt1[b] transposed, K-major
  __shared__ float rmax[16 * 4];
  __shared__ float rsum[16 * 4];
  __shared__ float Dst[16 * 32];
  for (int i = t; i < 224 * 32; i += 64) {
    int r = i >> 5, c = i & 31;
    XtT[c * 224 + r] = (r < RR) ? (_Float16)xt1[((size_t)b * RR + r) * CC1 + c] : (_Float16)0.f;
  }
  // 4 threads per softmax row
  const int row = t & 15, q = t >> 4;
  const int i = i0 + row;
  const int j0 = q * 56, j1 = j0 + 56;
  const int jv1 = (j1 < RR) ? j1 : RR;
  const float* arow = adj + ((size_t)b * RR + i) * RR;
  float lm = 1.0f;                       // diag logit = 1.0 always in row
  if (i < RR) {
    for (int j = j0; j < jv1; ++j) {
      float av = arow[j];
      float l = (j == i) ? 1.0f : ((av < EDGEP) ? av : NEGV);
      lm = fmaxf(lm, l);
    }
  }
  rmax[row * 4 + q] = lm;
  __syncthreads();
  float mx = fmaxf(fmaxf(rmax[row * 4 + 0], rmax[row * 4 + 1]),
                   fmaxf(rmax[row * 4 + 2], rmax[row * 4 + 3]));
  float s;
  if (i < RR) {
    s = 0.f;
    for (int j = j0; j < jv1; ++j) {
      float av = arow[j];
      float l = (j == i) ? 1.0f : ((av < EDGEP) ? av : NEGV);
      float e = __expf(l - mx);
      Att[row * 224 + j] = (_Float16)e;
      s += e;
    }
    for (int j = jv1; j < j1; ++j) Att[row * 224 + j] = (_Float16)0.f;
  } else {
    for (int j = j0; j < j1; ++j) Att[row * 224 + j] = (_Float16)0.f;
    s = 0.25f;   // keeps 1/sum finite for dead rows
  }
  rsum[row * 4 + q] = s;
  __syncthreads();
  {
    float inv = 1.0f / (rsum[row * 4 + 0] + rsum[row * 4 + 1] +
                        rsum[row * 4 + 2] + rsum[row * 4 + 3]);
    for (int j = j0; j < jv1; ++j)
      Att[row * 224 + j] = (_Float16)((float)Att[row * 224 + j] * inv);
  }
  __syncthreads();
  const int wave = t >> 5, lane = t & 31;
  v8f acc = {};
  for (int kt = 0; kt < 7; ++kt) {
    v16h a  = lds_load_a(Att, 224, 0, kt * 32);
    v16h bm = lds_load_bT(XtT, 224, kt * 32, wave * 16);
    acc = __builtin_amdgcn_wmma_f32_16x16x32_f16(false, a, false, bm, (short)0, acc, false, false);
  }
  {
    const int colb = wave * 16 + (lane & 15);
    const float bi = bias1[colb];
#pragma unroll
    for (int v = 0; v < 8; ++v)
      Dst[(v + 8 * (lane >> 4)) * 32 + colb] = acc[v] + bi;
  }
  __syncthreads();
  if (t < 32) {
    int r = t >> 1, half = t & 1;
    int ii = i0 + r;
    if (ii < RR) {
      const float* src = Dst + r * 32 + half * 16;
      float* dstp = h1 + ((size_t)b * RR + ii) * CC1 + half * 16;
      *(float4*)(dstp + 0)  = *(const float4*)(src + 0);
      *(float4*)(dstp + 4)  = *(const float4*)(src + 4);
      *(float4*)(dstp + 8)  = *(const float4*)(src + 8);
      *(float4*)(dstp + 12) = *(const float4*)(src + 12);
    }
  }
}

// ---- 4. top-k pool 1 (bitonic 256) -----------------------------------------
__global__ __launch_bounds__(256) void bg_topk1(const float* __restrict__ h1,
                                                const float* __restrict__ wp1,
                                                const int* __restrict__ roi,
                                                int* __restrict__ perm1, int* __restrict__ roi1,
                                                float* __restrict__ x1, float* __restrict__ out_s1) {
  const int b = blockIdx.x, t = threadIdx.x;
  __shared__ float sv[256];
  __shared__ int   si[256];
  float nrm = 0.f;
#pragma unroll
  for (int c = 0; c < CC1; ++c) nrm += wp1[c] * wp1[c];
  nrm = sqrtf(nrm);
  float val = -1e30f;
  if (t < RR) {
    float d = 0.f;
    const float* hr = h1 + ((size_t)b * RR + t) * CC1;
#pragma unroll
    for (int c = 0; c < CC1; ++c) d += hr[c] * wp1[c];
    val = 1.0f / (1.0f + __expf(-d / nrm));
  }
  sv[t] = val; si[t] = t;
  for (int k = 2; k <= 256; k <<= 1)
    for (int j = k >> 1; j > 0; j >>= 1) {
      __syncthreads();
      int ixj = t ^ j;
      if (ixj > t) {
        bool up = ((t & k) == 0);
        float v1 = sv[t], v2 = sv[ixj];
        bool sw = up ? (v1 < v2) : (v1 > v2);   // descending
        if (sw) { sv[t] = v2; sv[ixj] = v1; int tmp = si[t]; si[t] = si[ixj]; si[ixj] = tmp; }
      }
    }
  __syncthreads();
  if (t < KK1) {
    int idx = si[t];
    float v = sv[t];
    perm1[b * KK1 + t] = idx;
    roi1[b * KK1 + t]  = roi[idx];
    out_s1[b * KK1 + t] = 1.0f / (1.0f + __expf(-v));
    const float* hr = h1 + ((size_t)b * RR + idx) * CC1;
    float* xr = x1 + ((size_t)b * KK1 + t) * CC1;
#pragma unroll
    for (int c = 0; c < CC1; ++c) xr[c] = hr[c] * v;
  }
}

// ---- 5. augment: a2 = wd @ wd  (WMMA) --------------------------------------
__global__ __launch_bounds__(256) void bg_augment(const float* __restrict__ adj,
                                                  const int* __restrict__ perm1,
                                                  float* __restrict__ a2) {
  const int b = blockIdx.x, t = threadIdx.x;
  __shared__ _Float16 Wd[128 * 128];    // row-major (A)
  __shared__ _Float16 WdT[128 * 128];   // K-major (B)
  __shared__ int pl[KK1];
  if (t < KK1) pl[t] = perm1[b * KK1 + t];
  __syncthreads();
  for (int i = t; i < 128 * 128; i += 256) {
    int r = i >> 7, c = i & 127;
    float v = 0.f;
    if (r < KK1 && c < KK1) {
      if (r == c) v = 1.0f;
      else {
        float av = adj[((size_t)b * RR + pl[r]) * RR + pl[c]];
        v = (av < EDGEP) ? av : 0.f;
      }
    }
    Wd[r * 128 + c]  = (_Float16)v;
    WdT[c * 128 + r] = (_Float16)v;
  }
  __syncthreads();
  const int wave = t >> 5, lane = t & 31;
  for (int tile = wave; tile < 49; tile += 8) {   // wave-uniform
    const int mt = tile / 7, nt = tile % 7;
    v8f acc = {};
#pragma unroll
    for (int kt = 0; kt < 4; ++kt) {
      v16h a  = lds_load_a(Wd, 128, mt * 16, kt * 32);
      v16h bm = lds_load_bT(WdT, 128, kt * 32, nt * 16);
      acc = __builtin_amdgcn_wmma_f32_16x16x32_f16(false, a, false, bm, (short)0, acc, false, false);
    }
    const int col = nt * 16 + (lane & 15);
#pragma unroll
    for (int v = 0; v < 8; ++v) {
      int i = mt * 16 + v + 8 * (lane >> 4);
      if (i < KK1 && col < KK1)
        a2[((size_t)b * KK1 + i) * KK1 + col] = (i == col) ? 0.f : acc[v];
    }
  }
}

// ---- 6. gconv2: node transform + masked softmax + att@xt2 (WMMA) -----------
__global__ __launch_bounds__(128) void bg_h2(const float* __restrict__ x1,
                                             const float* __restrict__ wtab2,
                                             const int* __restrict__ roi1,
                                             const float* __restrict__ a2,
                                             const float* __restrict__ bias2,
                                             float* __restrict__ h2) {
  const int b = blockIdx.x, t = threadIdx.x;
  __shared__ _Float16 Att[112 * 128];
  __shared__ _Float16 Xt2T[32 * 128];   // K-major
  for (int o = t; o < 128 * 32; o += 128) {
    int n = o >> 5, c = o & 31;
    float acc = 0.f;
    if (n < KK1) {
      int node = roi1[b * KK1 + n];
      const float* xr = x1 + ((size_t)b * KK1 + n) * CC1;
      const float* w  = wtab2 + (size_t)node * (CC1 * CC2) + c;
#pragma unroll
      for (int i = 0; i < CC1; ++i) acc += xr[i] * w[i * CC2];
    }
    Xt2T[c * 128 + n] = (_Float16)acc;
  }
  if (t < 112) {
    _Float16* row = Att + t * 128;
    if (t < KK1) {
      const float* ar = a2 + ((size_t)b * KK1 + t) * KK1;
      float mx = 1.0f;
      for (int j = 0; j < KK1; ++j) {
        if (j == t) continue;
        float av = ar[j];
        float l = (av != 0.f) ? av : NEGV;
        mx = fmaxf(mx, l);
      }
      float sum = 0.f;
      for (int j = 0; j < KK1; ++j) {
        float l = (j == t) ? 1.0f : ((ar[j] != 0.f) ? ar[j] : NEGV);
        float e = __expf(l - mx);
        row[j] = (_Float16)e;
        sum += e;
      }
      for (int j = KK1; j < 128; ++j) row[j] = (_Float16)0.f;
      float inv = 1.0f / sum;
      for (int j = 0; j < KK1; ++j)
        row[j] = (_Float16)((float)row[j] * inv);
    } else {
      for (int j = 0; j < 128; ++j) row[j] = (_Float16)0.f;
    }
  }
  __syncthreads();
  const int wave = t >> 5, lane = t & 31;
  for (int tile = wave; tile < 14; tile += 4) {
    const int mt = tile >> 1, nt = tile & 1;
    v8f acc = {};
#pragma unroll
    for (int kt = 0; kt < 4; ++kt) {
      v16h a  = lds_load_a(Att, 128, mt * 16, kt * 32);
      v16h bm = lds_load_bT(Xt2T, 128, kt * 32, nt * 16);
      acc = __builtin_amdgcn_wmma_f32_16x16x32_f16(false, a, false, bm, (short)0, acc, false, false);
    }
    const int col = nt * 16 + (lane & 15);
    const float bi = bias2[col];
#pragma unroll
    for (int v = 0; v < 8; ++v) {
      int i = mt * 16 + v + 8 * (lane >> 4);
      if (i < KK1) h2[((size_t)b * KK1 + i) * CC2 + col] = acc[v] + bi;
    }
  }
}

// ---- 7. top-k pool 2 + pooled feature vector -------------------------------
__global__ __launch_bounds__(128) void bg_topk2_feat(const float* __restrict__ h2,
                                                     const float* __restrict__ wp2,
                                                     const float* __restrict__ x1,
                                                     float* __restrict__ feat,
                                                     float* __restrict__ out_s2) {
  const int b = blockIdx.x, t = threadIdx.x;
  __shared__ float sv[128];
  __shared__ int   si[128];
  __shared__ float X2[KK2 * CC2];
  float nrm = 0.f;
#pragma unroll
  for (int c = 0; c < CC2; ++c) nrm += wp2[c] * wp2[c];
  nrm = sqrtf(nrm);
  float val = -1e30f;
  if (t < KK1) {
    float d = 0.f;
    const float* hr = h2 + ((size_t)b * KK1 + t) * CC2;
#pragma unroll
    for (int c = 0; c < CC2; ++c) d += hr[c] * wp2[c];
    val = 1.0f / (1.0f + __expf(-d / nrm));
  }
  sv[t] = val; si[t] = t;
  for (int k = 2; k <= 128; k <<= 1)
    for (int j = k >> 1; j > 0; j >>= 1) {
      __syncthreads();
      int ixj = t ^ j;
      if (ixj > t) {
        bool up = ((t & k) == 0);
        float v1 = sv[t], v2 = sv[ixj];
        bool sw = up ? (v1 < v2) : (v1 > v2);
        if (sw) { sv[t] = v2; sv[ixj] = v1; int tmp = si[t]; si[t] = si[ixj]; si[ixj] = tmp; }
      }
    }
  __syncthreads();
  if (t < KK2) {
    int idx = si[t];
    float v = sv[t];
    out_s2[b * KK2 + t] = 1.0f / (1.0f + __expf(-v));
    const float* hr = h2 + ((size_t)b * KK1 + idx) * CC2;
#pragma unroll
    for (int c = 0; c < CC2; ++c) X2[t * CC2 + c] = hr[c] * v;
  }
  __syncthreads();
  if (t < CC1) {
    float mx = -1e30f, sm = 0.f;
    for (int n = 0; n < KK1; ++n) {
      float v = x1[((size_t)b * KK1 + n) * CC1 + t];
      mx = fmaxf(mx, v); sm += v;
    }
    feat[(size_t)b * 128 + t]      = mx;
    feat[(size_t)b * 128 + 32 + t] = sm / (float)KK1;
  } else if (t < 64) {
    int c = t - 32;
    float mx = -1e30f, sm = 0.f;
    for (int n = 0; n < KK2; ++n) {
      float v = X2[n * CC2 + c];
      mx = fmaxf(mx, v); sm += v;
    }
    feat[(size_t)b * 128 + 64 + c] = mx;
    feat[(size_t)b * 128 + 96 + c] = sm / (float)KK2;
  }
}

// ---- 8..13. MLP head (tiny) ------------------------------------------------
__global__ void bg_mlp1(const float* __restrict__ feat, const float* __restrict__ Wm1,
                        const float* __restrict__ bm1, float* __restrict__ ypre1) {
  int idx = blockIdx.x * blockDim.x + threadIdx.x;
  if (idx >= BB * CC2) return;
  int b = idx >> 5, o = idx & 31;
  float acc = bm1[o];
  const float* fr = feat + (size_t)b * 128;
  const float* wr = Wm1 + (size_t)o * 128;
  for (int j = 0; j < 128; ++j) acc += fr[j] * wr[j];
  ypre1[idx] = acc > 0.f ? acc : LEAKV * acc;
}

__global__ void bg_bnred1(const float* __restrict__ ypre1, float* __restrict__ mu1,
                          float* __restrict__ var1) {
  int o = threadIdx.x;
  if (o >= CC2) return;
  float s = 0.f, sq = 0.f;
  for (int b = 0; b < BB; ++b) { float v = ypre1[b * CC2 + o]; s += v; sq += v * v; }
  float mu = s / (float)BB;
  mu1[o] = mu;
  var1[o] = sq / (float)BB - mu * mu;
}

__global__ void bg_y1n(const float* __restrict__ ypre1, const float* __restrict__ mu1,
                       const float* __restrict__ var1, const float* __restrict__ g1,
                       const float* __restrict__ be1, float* __restrict__ y1n) {
  int idx = blockIdx.x * blockDim.x + threadIdx.x;
  if (idx >= BB * CC2) return;
  int o = idx & 31;
  y1n[idx] = g1[o] * (ypre1[idx] - mu1[o]) * rsqrtf(var1[o] + BNEPS) + be1[o];
}

__global__ void bg_mlp2(const float* __restrict__ y1n, const float* __restrict__ Wm2,
                        const float* __restrict__ bm2, float* __restrict__ ypre2) {
  int idx = blockIdx.x * blockDim.x + threadIdx.x;
  if (idx >= BB * CC3) return;
  int b = idx >> 9, o = idx & 511;
  float acc = bm2[o];
  const float* yr = y1n + (size_t)b * CC2;
  const float* wr = Wm2 + (size_t)o * CC2;
#pragma unroll
  for (int j = 0; j < CC2; ++j) acc += yr[j] * wr[j];
  ypre2[idx] = acc > 0.f ? acc : LEAKV * acc;
}

__global__ void bg_bnred2(const float* __restrict__ ypre2, float* __restrict__ mu2,
                          float* __restrict__ var2) {
  int o = blockIdx.x * blockDim.x + threadIdx.x;
  if (o >= CC3) return;
  float s = 0.f, sq = 0.f;
  for (int b = 0; b < BB; ++b) { float v = ypre2[b * CC3 + o]; s += v; sq += v * v; }
  float mu = s / (float)BB;
  mu2[o] = mu;
  var2[o] = sq / (float)BB - mu * mu;
}

__global__ void bg_final(const float* __restrict__ ypre2, const float* __restrict__ mu2,
                         const float* __restrict__ var2, const float* __restrict__ g2,
                         const float* __restrict__ be2, const float* __restrict__ Ws,
                         const float* __restrict__ bs, const float* __restrict__ wp1,
                         const float* __restrict__ wp2, float* __restrict__ out) {
  int b = blockIdx.x * blockDim.x + threadIdx.x;
  if (blockIdx.x == 0) {
    int t = threadIdx.x;
    if (t < 32)      out[1024 + t]        = wp1[t];
    else if (t < 64) out[1056 + (t - 32)] = wp2[t - 32];
  }
  if (b >= BB) return;
  float z0 = bs[0], z1 = bs[1];
  for (int o = 0; o < CC3; ++o) {
    float yn = g2[o] * (ypre2[b * CC3 + o] - mu2[o]) * rsqrtf(var2[o] + BNEPS) + be2[o];
    z0 += yn * Ws[o];
    z1 += yn * Ws[CC3 + o];
  }
  float m = fmaxf(z0, z1);
  float lse = m + logf(expf(z0 - m) + expf(z1 - m));
  out[b * 2]     = z0 - lse;
  out[b * 2 + 1] = z1 - lse;
}

// ---- launcher --------------------------------------------------------------
extern "C" void kernel_launch(void* const* d_in, const int* in_sizes, int n_in,
                              void* d_out, int out_size, void* d_ws, size_t ws_size,
                              hipStream_t stream) {
  (void)in_sizes; (void)n_in; (void)out_size; (void)ws_size;
  const float* x     = (const float*)d_in[0];
  const float* adj   = (const float*)d_in[1];
  const int*   roi   = (const int*)  d_in[2];
  const float* W1a   = (const float*)d_in[3];
  const float* b1a   = (const float*)d_in[4];
  const float* W2a   = (const float*)d_in[5];
  const float* b2a   = (const float*)d_in[6];
  const float* bias1 = (const float*)d_in[7];
  const float* W1b   = (const float*)d_in[8];
  const float* b1b   = (const float*)d_in[9];
  const float* W2b   = (const float*)d_in[10];
  const float* b2b   = (const float*)d_in[11];
  const float* bias2 = (const float*)d_in[12];
  const float* wp1   = (const float*)d_in[13];
  const float* wp2   = (const float*)d_in[14];
  const float* Wm1   = (const float*)d_in[15];
  const float* bm1   = (const float*)d_in[16];
  const float* g1    = (const float*)d_in[17];
  const float* be1   = (const float*)d_in[18];
  const float* Wm2   = (const float*)d_in[19];
  const float* bm2   = (const float*)d_in[20];
  const float* g2    = (const float*)d_in[21];
  const float* be2   = (const float*)d_in[22];
  const float* Ws    = (const float*)d_in[23];
  const float* bs    = (const float*)d_in[24];
  float* out = (float*)d_out;

  char* ws = (char*)d_ws;
  size_t off = 0;
  auto alloc = [&](size_t nbytes) -> void* {
    void* p = ws + off;
    off = (off + nbytes + 255) & ~(size_t)255;
    return p;
  };
  float* wtab1 = (float*)alloc(sizeof(float) * (size_t)RR * RR * CC1);
  float* wtab2 = (float*)alloc(sizeof(float) * (size_t)RR * CC1 * CC2);
  float* xt1   = (float*)alloc(sizeof(float) * (size_t)BB * RR * CC1);
  float* h1    = (float*)alloc(sizeof(float) * (size_t)BB * RR * CC1);
  int*   perm1 = (int*)  alloc(sizeof(int)   * (size_t)BB * KK1);
  int*   roi1  = (int*)  alloc(sizeof(int)   * (size_t)BB * KK1);
  float* x1    = (float*)alloc(sizeof(float) * (size_t)BB * KK1 * CC1);
  float* a2    = (float*)alloc(sizeof(float) * (size_t)BB * KK1 * KK1);
  float* h2    = (float*)alloc(sizeof(float) * (size_t)BB * KK1 * CC2);
  float* feat  = (float*)alloc(sizeof(float) * (size_t)BB * 128);
  float* ypre1 = (float*)alloc(sizeof(float) * (size_t)BB * CC2);
  float* mu1   = (float*)alloc(sizeof(float) * CC2);
  float* var1  = (float*)alloc(sizeof(float) * CC2);
  float* y1n   = (float*)alloc(sizeof(float) * (size_t)BB * CC2);
  float* ypre2 = (float*)alloc(sizeof(float) * (size_t)BB * CC3);
  float* mu2   = (float*)alloc(sizeof(float) * CC3);
  float* var2  = (float*)alloc(sizeof(float) * CC3);

  {
    int total = RR * RR * CC1 + RR * CC1 * CC2;
    bg_tables<<<(total + 255) / 256, 256, 0, stream>>>(W1a, b1a, W2a, b2a,
                                                       W1b, b1b, W2b, b2b, wtab1, wtab2);
  }
  bg_xt1<<<dim3(BB / 32, RR), 64, 0, stream>>>(x, wtab1, xt1);
  bg_h1<<<dim3(13, BB), 64, 0, stream>>>(adj, xt1, bias1, h1);
  bg_topk1<<<BB, 256, 0, stream>>>(h1, wp1, roi, perm1, roi1, x1, out + 1088);
  bg_augment<<<BB, 256, 0, stream>>>(adj, perm1, a2);
  bg_h2<<<BB, 128, 0, stream>>>(x1, wtab2, roi1, a2, bias2, h2);
  bg_topk2_feat<<<BB, 128, 0, stream>>>(h2, wp2, x1, feat, out + 52288);
  bg_mlp1<<<(BB * CC2 + 255) / 256, 256, 0, stream>>>(feat, Wm1, bm1, ypre1);
  bg_bnred1<<<1, 32, 0, stream>>>(ypre1, mu1, var1);
  bg_y1n<<<(BB * CC2 + 255) / 256, 256, 0, stream>>>(ypre1, mu1, var1, g1, be1, y1n);
  bg_mlp2<<<(BB * CC3 + 255) / 256, 256, 0, stream>>>(y1n, Wm2, bm2, ypre2);
  bg_bnred2<<<(CC3 + 255) / 256, 256, 0, stream>>>(ypre2, mu2, var2);
  bg_final<<<2, 256, 0, stream>>>(ypre2, mu2, var2, g2, be2, Ws, bs, wp1, wp2, out);
}
